// GATv2_GRU_Model_18537078849862
// MI455X (gfx1250) — matrix operand order repaired and checked by reading, not compiled, not run on hardware
//
#include <hip/hip_runtime.h>
#include <hip/hip_bf16.h>
#include <math.h>

// ---------------- problem constants ----------------
#define Bsz   32
#define Tsz   256
#define Nsz   207
#define Fsz   1
#define HGAT  8
#define Esz   1722
#define Gsz   (Bsz * Tsz)              // 8192 graphs
#define TOT   (Gsz * Nsz)              // 1,695,744 nodes
#define NEDGE ((size_t)Gsz * Esz + TOT)// 15,802,368 edge instances
#define Hgru  (HGAT * Nsz)             // 1656
#define G3    (3 * Hgru)               // 4968
#define KPAD  1664                     // 52 * 32
#define NPAD  4992                     // 78 * 64
#define NEG_SLOPE 0.2f

typedef __bf16 bf16;
typedef __attribute__((ext_vector_type(16))) __bf16 v16bf;
typedef __attribute__((ext_vector_type(8)))  __bf16 v8bf;
typedef __attribute__((ext_vector_type(8)))  float  v8f;

union Frag16 { v16bf v; v8bf h[2]; };

// ---------------- helpers ----------------
__device__ __forceinline__ unsigned short f32_to_bf16_rne(float f) {
    unsigned u = __float_as_uint(f);
    unsigned r = u + 0x7FFFu + ((u >> 16) & 1u);
    return (unsigned short)(r >> 16);
}

__device__ __forceinline__ void atomicMaxF(float* addr, float val) {
    if (val >= 0.0f) atomicMax((int*)addr, __float_as_int(val));
    else             atomicMin((unsigned int*)addr, __float_as_uint(val));
}

// ---------------- GAT kernels ----------------
__global__ void fill_f32(float* p, float v, size_t n) {
    size_t i = (size_t)blockIdx.x * blockDim.x + threadIdx.x;
    if (i < n) p[i] = v;
}

// xl = x @ Wl^T + bl ; xr = x @ Wr^T + br   (per node, 8 outputs each)
__global__ void gat_node_linear(const float* __restrict__ x, int Fin,
                                const float* __restrict__ Wl, const float* __restrict__ bl,
                                const float* __restrict__ Wr, const float* __restrict__ br,
                                float* __restrict__ xl, float* __restrict__ xr) {
    int nd = blockIdx.x * blockDim.x + threadIdx.x;
    if (nd >= TOT) return;
    float xi[8];
    for (int k = 0; k < Fin; ++k) xi[k] = x[(size_t)nd * Fin + k];
    for (int j = 0; j < HGAT; ++j) {
        float sl = bl[j], sr = br[j];
        for (int k = 0; k < Fin; ++k) {
            sl += xi[k] * Wl[j * Fin + k];
            sr += xi[k] * Wr[j * Fin + k];
        }
        xl[(size_t)nd * HGAT + j] = sl;
        xr[(size_t)nd * HGAT + j] = sr;
    }
}

__device__ __forceinline__ void edge_endpoints(size_t gid, const int* ei, int& s, int& d) {
    if (gid < (size_t)Gsz * Esz) {
        int g = (int)(gid / Esz);
        int e = (int)(gid - (size_t)g * Esz);
        s = ei[e] + g * Nsz;
        d = ei[Esz + e] + g * Nsz;
    } else {
        s = d = (int)(gid - (size_t)Gsz * Esz);   // self loop
    }
}

__global__ void gat_edge_score(const float* __restrict__ xl, const float* __restrict__ xr,
                               const float* __restrict__ att, const int* __restrict__ ei,
                               float* __restrict__ scores, float* __restrict__ smax) {
    size_t gid = (size_t)blockIdx.x * blockDim.x + threadIdx.x;
    if (gid >= NEDGE) return;
    int s, d; edge_endpoints(gid, ei, s, d);
    float sc = 0.0f;
#pragma unroll
    for (int j = 0; j < HGAT; ++j) {
        float v = xl[(size_t)s * HGAT + j] + xr[(size_t)d * HGAT + j];
        v = (v > 0.0f) ? v : NEG_SLOPE * v;
        sc += v * att[j];
    }
    scores[gid] = sc;
    atomicMaxF(&smax[d], sc);
}

__global__ void gat_edge_accum(const float* __restrict__ xl,
                               const int* __restrict__ ei,
                               const float* __restrict__ scores, const float* __restrict__ smax,
                               float* __restrict__ denom, float* __restrict__ accum) {
    size_t gid = (size_t)blockIdx.x * blockDim.x + threadIdx.x;
    if (gid >= NEDGE) return;
    int s, d; edge_endpoints(gid, ei, s, d);
    float ex = __expf(scores[gid] - smax[d]);
    atomicAdd(&denom[d], ex);
#pragma unroll
    for (int j = 0; j < HGAT; ++j)
        atomicAdd(&accum[(size_t)d * HGAT + j], ex * xl[(size_t)s * HGAT + j]);
}

__global__ void gat_finalize(const float* __restrict__ accum, const float* __restrict__ denom,
                             const float* __restrict__ bias, float* __restrict__ out) {
    size_t gid = (size_t)blockIdx.x * blockDim.x + threadIdx.x;
    if (gid >= (size_t)TOT * HGAT) return;
    size_t nd = gid >> 3;
    int j = (int)(gid & 7);
    out[gid] = accum[gid] / (denom[nd] + 1e-16f) + bias[j];
}

// ---------------- pack / convert kernels ----------------
__global__ void cvt_weight_bf16(const float* __restrict__ W, unsigned short* __restrict__ Wb) {
    size_t gid = (size_t)blockIdx.x * blockDim.x + threadIdx.x;
    if (gid >= (size_t)NPAD * KPAD) return;
    int r = (int)(gid / KPAD), c = (int)(gid % KPAD);
    Wb[gid] = (r < G3 && c < Hgru) ? f32_to_bf16_rne(W[(size_t)r * Hgru + c]) : (unsigned short)0;
}

__global__ void cvt_seq_bf16(const float* __restrict__ src, unsigned short* __restrict__ A) {
    size_t gid = (size_t)blockIdx.x * blockDim.x + threadIdx.x;
    if (gid >= (size_t)Gsz * Hgru) return;
    size_t row = gid / Hgru, c = gid % Hgru;
    A[row * KPAD + c] = f32_to_bf16_rne(src[gid]);
}

// ---------------- WMMA bf16 GEMM:  C = A(MxKPAD) @ W(NPADxKPAD)^T ----------------
// Wave computes (MT*16) x 64 tile; block = 4 waves stacked in M.
// grid = (ceil(M/(64*MT)), NPAD/64, kChunks).  kChunks>1 => split-K, fp32 atomic accumulate.
template<int MT, bool ATOMIC>
__global__ void wmma_gemm_bf16(const unsigned short* __restrict__ Aus,
                               const unsigned short* __restrict__ Wus,
                               float* __restrict__ C, int M, int Nreal) {
    const bf16* A = reinterpret_cast<const bf16*>(Aus);
    const bf16* W = reinterpret_cast<const bf16*>(Wus);
    int lane  = threadIdx.x & 31;
    int wave  = threadIdx.x >> 5;
    int tileM = (blockIdx.x * 4 + wave) * (16 * MT);
    if (tileM >= M) return;                 // wave-uniform; active waves keep EXEC all-ones
    int tileN = blockIdx.y * 64;
    int lh  = lane & 15;
    bool hi = lane >= 16;
    // split-K range (KPAD multiple of 32*gridDim.z by construction)
    int kcLen = KPAD / (int)gridDim.z;
    int kBeg  = (int)blockIdx.z * kcLen;
    int kEnd  = kBeg + kcLen;

    v8f acc[MT][4];
#pragma unroll
    for (int m = 0; m < MT; ++m)
#pragma unroll
        for (int s = 0; s < 4; ++s)
            acc[m][s] = (v8f){0.f,0.f,0.f,0.f,0.f,0.f,0.f,0.f};

    size_t arow[MT];
#pragma unroll
    for (int m = 0; m < MT; ++m) arow[m] = (size_t)(tileM + m * 16 + lh) * KPAD;

    for (int k0 = kBeg; k0 < kEnd; k0 += 32) {
        // A fragment 16x32: lanes 0-15 hold K {0..7,16..23}, lanes 16-31 hold K {8..15,24..31}
        Frag16 a[MT];
#pragma unroll
        for (int m = 0; m < MT; ++m) {
            const bf16* ap = A + arow[m] + k0 + (hi ? 8 : 0);
            a[m].h[0] = *(const v8bf*)(ap);
            a[m].h[1] = *(const v8bf*)(ap + 16);
        }
#pragma unroll
        for (int s = 0; s < 4; ++s) {
            // B fragment 32x16: lane n (0-15) = column n, K {0..7,16..23} ; lanes 16-31 = column, K {8..15,24..31}
            int col = tileN + s * 16 + lh;
            const bf16* bp = W + (size_t)col * KPAD + k0 + (hi ? 8 : 0);
            Frag16 b;
            b.h[0] = *(const v8bf*)(bp);
            b.h[1] = *(const v8bf*)(bp + 16);
#pragma unroll
            for (int m = 0; m < MT; ++m)
                acc[m][s] = __builtin_amdgcn_wmma_f32_16x16x32_bf16(
                    false, a[m].v, false, b.v, (short)0, acc[m][s], false, false);
        }
    }
    // C layout: VGPR r -> (row = base + r + (hi?8:0), col = tileN + s*16 + lh)
#pragma unroll
    for (int m = 0; m < MT; ++m) {
#pragma unroll
        for (int s = 0; s < 4; ++s) {
            int cc = tileN + s * 16 + lh;
            if (cc >= Nreal) continue;
#pragma unroll
            for (int r = 0; r < 8; ++r) {
                int row = tileM + m * 16 + r + (hi ? 8 : 0);
                if (ATOMIC) atomicAdd(&C[(size_t)row * NPAD + cc], acc[m][s][r]);
                else        C[(size_t)row * NPAD + cc] = acc[m][s][r];
            }
        }
    }
}

// ---------------- GRU gate (fused elementwise step) ----------------
__global__ void gru_gate(const float* __restrict__ xg, const float* __restrict__ hg,
                         const float* __restrict__ bih, const float* __restrict__ bhh,
                         float* __restrict__ h, unsigned short* __restrict__ hbf,
                         unsigned short* __restrict__ ybf, int t) {
    int gid = blockIdx.x * blockDim.x + threadIdx.x;
    if (gid >= Bsz * Hgru) return;
    int b = gid / Hgru, j = gid % Hgru;
    size_t row = (size_t)b * Tsz + t;
    float xr = xg[row * NPAD + j]            + bih[j];
    float xz = xg[row * NPAD + Hgru + j]     + bih[Hgru + j];
    float xn = xg[row * NPAD + 2 * Hgru + j] + bih[2 * Hgru + j];
    float hr = hg[(size_t)b * NPAD + j]            + bhh[j];
    float hz = hg[(size_t)b * NPAD + Hgru + j]     + bhh[Hgru + j];
    float hn = hg[(size_t)b * NPAD + 2 * Hgru + j] + bhh[2 * Hgru + j];
    float r = 1.0f / (1.0f + __expf(-(xr + hr)));
    float z = 1.0f / (1.0f + __expf(-(xz + hz)));
    float n = tanhf(xn + r * hn);
    float hprev = h[gid];
    float hnew  = (1.0f - z) * n + z * hprev;
    h[gid] = hnew;
    unsigned short hb = f32_to_bf16_rne(hnew);
    hbf[(size_t)b * KPAD + j] = hb;
    if (ybf) ybf[row * KPAD + j] = hb;
}

// ---------------- final FC ----------------
__global__ void fc_kernel(const float* __restrict__ h, const float* __restrict__ Wfc,
                          const float* __restrict__ bfc, float* __restrict__ out) {
    int gid = blockIdx.x * blockDim.x + threadIdx.x;
    if (gid >= Bsz * Nsz) return;
    int b = gid / Nsz, n = gid % Nsz;
    float s = bfc[n];
    const float* hr = h + (size_t)b * Hgru;
    const float* wr = Wfc + (size_t)n * Hgru;
    for (int k = 0; k < Hgru; ++k) s += hr[k] * wr[k];
    out[gid] = s;
}

// ---------------- host orchestration ----------------
static inline size_t align256(size_t x) { return (x + 255) & ~(size_t)255; }

extern "C" void kernel_launch(void* const* d_in, const int* in_sizes, int n_in,
                              void* d_out, int out_size, void* d_ws, size_t ws_size,
                              hipStream_t stream) {
    (void)in_sizes; (void)n_in; (void)out_size; (void)ws_size;
    const float* gseq = (const float*)d_in[0];
    const int*   ei   = (const int*)d_in[1];
    const float* Wl1  = (const float*)d_in[2];  const float* bl1 = (const float*)d_in[3];
    const float* Wr1  = (const float*)d_in[4];  const float* br1 = (const float*)d_in[5];
    const float* att1 = (const float*)d_in[6];  const float* bias1 = (const float*)d_in[7];
    const float* Wl2  = (const float*)d_in[8];  const float* bl2 = (const float*)d_in[9];
    const float* Wr2  = (const float*)d_in[10]; const float* br2 = (const float*)d_in[11];
    const float* att2 = (const float*)d_in[12]; const float* bias2 = (const float*)d_in[13];
    const float* Wih0 = (const float*)d_in[14]; const float* Whh0 = (const float*)d_in[15];
    const float* bih0 = (const float*)d_in[16]; const float* bhh0 = (const float*)d_in[17];
    const float* Wih1 = (const float*)d_in[18]; const float* Whh1 = (const float*)d_in[19];
    const float* bih1 = (const float*)d_in[20]; const float* bhh1 = (const float*)d_in[21];
    const float* Wfc  = (const float*)d_in[22]; const float* bfc = (const float*)d_in[23];
    float* out = (float*)d_out;
    char* ws = (char*)d_ws;

    // ---- persistent zone ----
    size_t off = 0;
    auto take = [&](size_t bytes) { size_t o = off; off = align256(off + bytes); return o; };
    const size_t szA   = (size_t)Gsz * KPAD * 2;            // bf16 activation matrix
    const size_t szW   = (size_t)NPAD * KPAD * 2;           // padded bf16 weight
    size_t oA0  = take(szA);
    size_t oA1  = take(szA);
    size_t oWb[4]; for (int i = 0; i < 4; ++i) oWb[i] = take(szW);
    size_t oH   = take((size_t)Bsz * Hgru * 4);
    size_t oHbf = take((size_t)Bsz * KPAD * 2);
    size_t oHg  = take((size_t)Bsz * NPAD * 4);
    // ---- scratch zone (GAT buffers; later reused for xg) ----
    size_t zS   = off;
    size_t oXl  = take((size_t)TOT * HGAT * 4);
    size_t oXr  = take((size_t)TOT * HGAT * 4);
    size_t oAcc = take((size_t)TOT * HGAT * 4);
    size_t oNA  = take((size_t)TOT * HGAT * 4);
    size_t oNB  = take((size_t)TOT * HGAT * 4);
    size_t oSc  = take(NEDGE * 4);
    size_t oSm  = take((size_t)TOT * 4);
    size_t oDen = take((size_t)TOT * 4);
    size_t oXg  = zS;                                       // reuse: GAT scratch dead by then

    const int TB = 256;
    dim3 gemmBlk(128);
    dim3 gridBig((Gsz / 32 + 3) / 4, NPAD / 64, 1);         // MT=2 : 64 x 78 blocks
    dim3 gridStep(1, NPAD / 64, 4);                         // MT=1 : split-K x4 (13 K-steps each)

    // ---- convert weights to padded bf16 ----
    {
        size_t n = (size_t)NPAD * KPAD;
        int gb = (int)((n + TB - 1) / TB);
        const float* srcs[4] = {Wih0, Whh0, Wih1, Whh1};
        for (int i = 0; i < 4; ++i)
            cvt_weight_bf16<<<gb, TB, 0, stream>>>(srcs[i], (unsigned short*)(ws + oWb[i]));
    }

    // ---- GAT layers ----
    auto run_gat = [&](const float* x, int Fin,
                       const float* Wl, const float* bl, const float* Wr, const float* br,
                       const float* att, const float* bias, float* dst) {
        fill_f32<<<(TOT + TB - 1) / TB, TB, 0, stream>>>((float*)(ws + oSm), -3.0e38f, TOT);
        hipMemsetAsync(ws + oDen, 0, (size_t)TOT * 4, stream);
        hipMemsetAsync(ws + oAcc, 0, (size_t)TOT * HGAT * 4, stream);
        gat_node_linear<<<(TOT + TB - 1) / TB, TB, 0, stream>>>(
            x, Fin, Wl, bl, Wr, br, (float*)(ws + oXl), (float*)(ws + oXr));
        int egrid = (int)((NEDGE + TB - 1) / TB);
        gat_edge_score<<<egrid, TB, 0, stream>>>(
            (float*)(ws + oXl), (float*)(ws + oXr), att, ei,
            (float*)(ws + oSc), (float*)(ws + oSm));
        gat_edge_accum<<<egrid, TB, 0, stream>>>(
            (float*)(ws + oXl), ei, (float*)(ws + oSc), (float*)(ws + oSm),
            (float*)(ws + oDen), (float*)(ws + oAcc));
        gat_finalize<<<(int)(((size_t)TOT * HGAT + TB - 1) / TB), TB, 0, stream>>>(
            (float*)(ws + oAcc), (float*)(ws + oDen), bias, dst);
    };
    run_gat(gseq, Fsz, Wl1, bl1, Wr1, br1, att1, bias1, (float*)(ws + oNA));
    run_gat((float*)(ws + oNA), HGAT, Wl2, bl2, Wr2, br2, att2, bias2, (float*)(ws + oNB));

    // ---- pack GAT output -> bf16 A0 (zero pad cols), zero A1 + h-state pads ----
    hipMemsetAsync(ws + oA0, 0, szA, stream);
    hipMemsetAsync(ws + oA1, 0, szA, stream);
    cvt_seq_bf16<<<(int)(((size_t)Gsz * Hgru + TB - 1) / TB), TB, 0, stream>>>(
        (float*)(ws + oNB), (unsigned short*)(ws + oA0));

    // ---- GRU layers ----
    auto run_gru = [&](size_t oAin, int wIh, int wHh,
                       const float* bih, const float* bhh, unsigned short* ybf) {
        // input projection: xg = A @ Wih^T   (compute-bound, MT=2 tiles)
        wmma_gemm_bf16<2, false><<<gridBig, gemmBlk, 0, stream>>>(
            (unsigned short*)(ws + oAin), (unsigned short*)(ws + oWb[wIh]),
            (float*)(ws + oXg), Gsz, G3);
        hipMemsetAsync(ws + oH, 0, (size_t)Bsz * Hgru * 4, stream);
        hipMemsetAsync(ws + oHbf, 0, (size_t)Bsz * KPAD * 2, stream);
        for (int t = 0; t < Tsz; ++t) {
            // recurrence: hg = h @ Whh^T  (latency-bound -> split-K atomic accumulate)
            hipMemsetAsync(ws + oHg, 0, (size_t)Bsz * NPAD * 4, stream);
            wmma_gemm_bf16<1, true><<<gridStep, gemmBlk, 0, stream>>>(
                (unsigned short*)(ws + oHbf), (unsigned short*)(ws + oWb[wHh]),
                (float*)(ws + oHg), Bsz, G3);
            gru_gate<<<(Bsz * Hgru + TB - 1) / TB, TB, 0, stream>>>(
                (float*)(ws + oXg), (float*)(ws + oHg), bih, bhh,
                (float*)(ws + oH), (unsigned short*)(ws + oHbf), ybf, t);
        }
    };
    run_gru(oA0, 0, 1, bih0, bhh0, (unsigned short*)(ws + oA1));   // layer 0 -> y into A1
    run_gru(oA1, 2, 3, bih1, bhh1, (unsigned short*)nullptr);      // layer 1 -> h only

    // ---- final FC ----
    fc_kernel<<<(Bsz * Nsz + TB - 1) / TB, TB, 0, stream>>>(
        (float*)(ws + oH), Wfc, bfc, out);
}
